// HPrior_VAE_78812649881853
// MI455X (gfx1250) — matrix operand order repaired and checked by reading, not compiled, not run on hardware
//
#include <hip/hip_runtime.h>
#include <hip/hip_bf16.h>
#include <math.h>

// ---------------------------------------------------------------------------
// HPrior VAE forward for MI455X (gfx1250, wave32, WMMA bf16 + async LDS copy)
//   B=512 T=200 V=3356 E=1128 D0=20000 H=1024 L=256
// Rewrite: einsum('bte,le->btl', embed[ids], Wi) == (embed @ Wi^T)[ids]
// Split-K on the occupancy-starved GEMMs (grid.z slices + combine kernel).
// ---------------------------------------------------------------------------

#define LATENT 256
#define B_     512
#define T_     200
#define V_     3356
#define VPAD_  3360     // V rounded up to 32
#define E_     1128
#define D0_    20000
#define H_     1024

#define LDS_PITCH 36    // floats per row in LDS tiles (bank-conflict-free pad)
#define A_BUF_BYTES (128 * LDS_PITCH * 4)
#define B_BUF_BYTES (64  * LDS_PITCH * 4)

typedef __attribute__((ext_vector_type(16))) __bf16 v16bf;
typedef __attribute__((ext_vector_type(8)))  float  v8f;
typedef __attribute__((ext_vector_type(4)))  float  v4f;

#define WMMA_BF16(a, b, c) \
  __builtin_amdgcn_wmma_f32_16x16x32_bf16(false, (a), false, (b), (short)0, (c), false, false)

#define WAIT_ASYNC(n) asm volatile("s_wait_asynccnt " #n ::: "memory")

// --- raw LDS reads (v4f chunks first, convert later -> batched ds_loads) ----
// A frag (16x32): lane<16: M=lane, K = kb+0..7 and kb+16..23 (kb=0);
//                 lane>=16: same M set, kb=8.
__device__ __forceinline__ void lds_read_a(const float* t, int row0, v4f q[4]) {
  const int lane = threadIdx.x & 31;
  const int kb = (lane >> 4) * 8;
  const float* p = t + (row0 + (lane & 15)) * LDS_PITCH + kb;
  q[0] = *(const v4f*)(p);      q[1] = *(const v4f*)(p + 4);
  q[2] = *(const v4f*)(p + 16); q[3] = *(const v4f*)(p + 20);
}
// B frag (32x16): lane<16: N=lane, K+0..7,16..23 ; lane>=16: N=lane-16, K+8..15,24..31
__device__ __forceinline__ void lds_read_b(const float* t, int row0, v4f q[4]) {
  const int lane = threadIdx.x & 31;
  const int koff = (lane >> 4) * 8;
  const float* p = t + (row0 + (lane & 15)) * LDS_PITCH + koff;
  q[0] = *(const v4f*)(p);     q[1] = *(const v4f*)(p + 4);
  q[2] = *(const v4f*)(p + 16); q[3] = *(const v4f*)(p + 20);
}
__device__ __forceinline__ v16bf cvt_frag(const v4f q[4]) {
  v16bf f;
#pragma unroll
  for (int j = 0; j < 4; ++j)
#pragma unroll
    for (int i = 0; i < 4; ++i) f[j * 4 + i] = (__bf16)q[j][i];
  return f;
}

// --- masked global-direct fragments (K tail only, gemmP) --------------------
__device__ __forceinline__ v16bf load_a_frag_tail(const float* __restrict__ A, int lda,
                                                  int row0, int mclamp, int k0, int K) {
  const int lane = threadIdx.x & 31;
  const int kb = (lane >> 4) * 8;
  int r = row0 + (lane & 15);
  if (r >= mclamp) r = mclamp - 1;
  const float* p = A + (size_t)r * lda;
  v16bf f;
#pragma unroll
  for (int i = 0; i < 16; ++i) {
    int k = k0 + kb + (i < 8 ? i : i + 8);
    f[i] = (k < K) ? (__bf16)p[k] : (__bf16)0.0f;
  }
  return f;
}
__device__ __forceinline__ v16bf load_b_frag_tail(const float* __restrict__ Bw, int ldb,
                                                  int col0, int nclamp, int k0, int K) {
  const int lane = threadIdx.x & 31;
  const int koff = (lane >> 4) * 8;
  int cc = col0 + (lane & 15);
  if (cc >= nclamp) cc = nclamp - 1;
  const float* p = Bw + (size_t)cc * ldb;
  v16bf f;
#pragma unroll
  for (int i = 0; i < 16; ++i) {
    int k = k0 + koff + (i < 8 ? i : i + 8);
    f[i] = (k < K) ? (__bf16)p[k] : (__bf16)0.0f;
  }
  return f;
}

// ---------------------------------------------------------------------------
// NT GEMM: C[M x N] = act( A[M x K] @ B[N x K]^T + bias[N] )   (f32 in/out,
// bf16 WMMA, f32 accum). 256 thr = 8 waves; wave tile 32x32 (2x2 WMMA);
// WG tile 128(M) x 64(N); operand tiles double-buffered in LDS via
// global_load_async_to_lds_b128 (ASYNCcnt pipeline).
// Split-K: gridDim.z slices of Kc (32-aligned); slice z covers
// [z*Kc, min(K,(z+1)*Kc)). If gridDim.z>1 raw partials are written to
// C + z*M*N (no bias/act); a combine kernel finishes. act: 0=none, 1=tanh.
// No early returns (barriers + EXEC all-ones for WMMA); stores guarded.
// ---------------------------------------------------------------------------
__global__ void __launch_bounds__(256)
wmma_gemm_nt(const float* __restrict__ A, const float* __restrict__ Bw,
             const float* __restrict__ bias, float* __restrict__ C,
             int M, int N, int K, int Kc, int Mstore, int act) {
  __shared__ __align__(16) float At[2][128 * LDS_PITCH];   // 2 x 18 KB
  __shared__ __align__(16) float Bt[2][64 * LDS_PITCH];    // 2 x  9 KB

  const int tid  = threadIdx.x;
  const int wid  = tid >> 5;
  const int lane = tid & 31;
  const int m0wg = blockIdx.y * 128;
  const int n0wg = blockIdx.x * 64;
  const int m0   = m0wg + (wid & 3) * 32;       // wave tile origin
  const int n0   = n0wg + (wid >> 2) * 32;

  const int kbeg = blockIdx.z * Kc;
  int kend = kbeg + Kc; if (kend > K) kend = K;
  const int klen  = kend - kbeg;
  const int kmain = klen & ~31;

  // ---- per-thread async-copy slots: A = 4 x 16B chunks, B = 2 x 16B chunks
  const float* gA[4]; unsigned lA0[4];
#pragma unroll
  for (int j = 0; j < 4; ++j) {
    const int c = j * 256 + tid, row = c >> 3, kc = (c & 7) * 4;  // kc in floats
    int gr = m0wg + row; if (gr >= M) gr = M - 1;
    gA[j] = A + (size_t)gr * K + kbeg + kc;
    lA0[j] = (unsigned)(uintptr_t)&At[0][row * LDS_PITCH + kc];
  }
  const float* gB[2]; unsigned lB0[2];
#pragma unroll
  for (int j = 0; j < 2; ++j) {
    const int c = j * 256 + tid, row = c >> 3, kc = (c & 7) * 4;
    int gc = n0wg + row; if (gc >= N) gc = N - 1;
    gB[j] = Bw + (size_t)gc * K + kbeg + kc;
    lB0[j] = (unsigned)(uintptr_t)&Bt[0][row * LDS_PITCH + kc];
  }

  // buffer select by address arithmetic (no dynamic VGPR indexing)
  auto issue_tile = [&](int buf, int k0) {
    const unsigned aoff = (unsigned)buf * (unsigned)A_BUF_BYTES;
    const unsigned boff = (unsigned)buf * (unsigned)B_BUF_BYTES;
#pragma unroll
    for (int j = 0; j < 4; ++j) {
      const float* g = gA[j] + k0;
      unsigned l = lA0[j] + aoff;
      asm volatile("global_load_async_to_lds_b128 %0, %1, off"
                   :: "v"(l), "v"(g) : "memory");
    }
#pragma unroll
    for (int j = 0; j < 2; ++j) {
      const float* g = gB[j] + k0;
      unsigned l = lB0[j] + boff;
      asm volatile("global_load_async_to_lds_b128 %0, %1, off"
                   :: "v"(l), "v"(g) : "memory");
    }
  };

  v8f acc[2][2] = {};
  const int am = (wid & 3) * 32;
  const int bn = (wid >> 2) * 32;

  int buf = 0;
  if (kmain > 0) issue_tile(0, 0);          // prologue: 6 async ops in flight
  for (int k0 = 0; k0 < kmain; k0 += 32) {
    const bool has_next = (k0 + 32 < kmain);
    if (has_next) issue_tile(buf ^ 1, k0 + 32);     // 12 in flight
    if (has_next) { WAIT_ASYNC(6); } else { WAIT_ASYNC(0); }
    __syncthreads();                        // all waves' current tile in LDS

    const float* As = &At[0][0] + buf * (128 * LDS_PITCH);
    const float* Bs = &Bt[0][0] + buf * (64 * LDS_PITCH);
    v4f ra0[4], ra1[4], rb0[4], rb1[4];     // all raw loads first ...
    lds_read_a(As, am,      ra0);
    lds_read_a(As, am + 16, ra1);
    lds_read_b(Bs, bn,      rb0);
    lds_read_b(Bs, bn + 16, rb1);
    v16bf a0 = cvt_frag(ra0), a1 = cvt_frag(ra1);   // ... then convert
    v16bf b0 = cvt_frag(rb0), b1 = cvt_frag(rb1);
    acc[0][0] = WMMA_BF16(a0, b0, acc[0][0]);
    acc[0][1] = WMMA_BF16(a0, b1, acc[0][1]);
    acc[1][0] = WMMA_BF16(a1, b0, acc[1][0]);
    acc[1][1] = WMMA_BF16(a1, b1, acc[1][1]);

    __syncthreads();                        // tile fully consumed before refill
    buf ^= 1;
  }

  if (kmain < klen) {  // K tail (only last slice when K % 32): direct global
    const int kt = kbeg + kmain;
    v16bf a0 = load_a_frag_tail(A, K, m0,      M, kt, K);
    v16bf a1 = load_a_frag_tail(A, K, m0 + 16, M, kt, K);
    v16bf b0 = load_b_frag_tail(Bw, K, n0,      N, kt, K);
    v16bf b1 = load_b_frag_tail(Bw, K, n0 + 16, N, kt, K);
    acc[0][0] = WMMA_BF16(a0, b0, acc[0][0]);
    acc[0][1] = WMMA_BF16(a0, b1, acc[0][1]);
    acc[1][0] = WMMA_BF16(a1, b0, acc[1][0]);
    acc[1][1] = WMMA_BF16(a1, b1, acc[1][1]);
  }

  // epilogue: C/D layout -> VGPR r holds M = r (lanes 0-15) or 8+r (lanes 16-31)
  const bool partial = (gridDim.z > 1);
  float* Cdst = C + (partial ? (size_t)blockIdx.z * Mstore * N : 0);
  const int cn = lane & 15;
  const int ro = (lane >> 4) * 8;
#pragma unroll
  for (int mt = 0; mt < 2; ++mt) {
#pragma unroll
    for (int nt = 0; nt < 2; ++nt) {
      const int col = n0 + nt * 16 + cn;
      if (col < N) {
        const float bv = partial ? 0.f : bias[col];
#pragma unroll
        for (int r = 0; r < 8; ++r) {
          const int row = m0 + mt * 16 + ro + r;
          if (row < Mstore) {
            float v = acc[mt][nt][r] + bv;
            if (act && !partial) v = tanhf(v);
            Cdst[(size_t)row * N + col] = v;
          }
        }
      }
    }
  }
}

// ---------------------------------------------------------------------------
// split-K combine: out[m,n] = act( sum_s part[s][m,n] + bias[n] )
// ---------------------------------------------------------------------------
__global__ void __launch_bounds__(256)
combine_kernel(const float* __restrict__ part, const float* __restrict__ bias,
               float* __restrict__ outp, int M, int N, int S, int act) {
  const int idx = blockIdx.x * 256 + threadIdx.x;
  if (idx >= M * N) return;
  const int col = idx % N;
  float v = bias[col];
  for (int s = 0; s < S; ++s) v += part[(size_t)s * M * N + idx];
  if (act) v = tanhf(v);
  outp[idx] = v;
}

// ---------------------------------------------------------------------------
// xdm[b,:] = x[b,:] / max(||x[b,:]||, 1e-12) * drop_mask[b,:]
// ---------------------------------------------------------------------------
__global__ void __launch_bounds__(256)
normalize_mask_kernel(const float* __restrict__ x, const float* __restrict__ mask,
                      float* __restrict__ xdm) {
  __shared__ float red[256];
  const int b = blockIdx.x;
  const float* xr = x + (size_t)b * D0_;
  float s = 0.f;
  for (int d = threadIdx.x; d < D0_; d += 256) { float v = xr[d]; s += v * v; }
  red[threadIdx.x] = s;
  __syncthreads();
  for (int o = 128; o > 0; o >>= 1) {
    if (threadIdx.x < o) red[threadIdx.x] += red[threadIdx.x + o];
    __syncthreads();
  }
  const float inv = 1.f / fmaxf(sqrtf(red[0]), 1e-12f);
  const float* mr = mask + (size_t)b * D0_;
  float* outp = xdm + (size_t)b * D0_;
  for (int d = threadIdx.x; d < D0_; d += 256) outp[d] = xr[d] * inv * mr[d];
}

// ---------------------------------------------------------------------------
// Per (b,l): mean/std(ddof=1) over T of P[ids[b,t], l]; prior = mean+sqrt(std)*eps
// ---------------------------------------------------------------------------
__global__ void __launch_bounds__(256)
stats_prior_kernel(const float* __restrict__ P, const int* __restrict__ ids,
                   const float* __restrict__ eps, float* __restrict__ prior) {
  __shared__ int ids_s[T_];
  const int b = blockIdx.x;
  for (int t = threadIdx.x; t < T_; t += 256) ids_s[t] = ids[b * T_ + t];
  __syncthreads();
  const int l = threadIdx.x;  // blockDim == LATENT == 256
  float s = 0.f, s2 = 0.f;
  for (int t = 0; t < T_; ++t) {
    const float v = P[(size_t)ids_s[t] * LATENT + l];
    s += v; s2 += v * v;
  }
  const float mean = s * (1.0f / T_);
  float var = (s2 - s * s * (1.0f / T_)) * (1.0f / (T_ - 1));
  var = fmaxf(var, 0.f);
  const float dev = sqrtf(var);                  // std (ddof=1)
  prior[b * LATENT + l] = mean + sqrtf(dev) * eps[b * LATENT + l];
}

// ---------------------------------------------------------------------------
// split gparams -> mu/logvar outputs; z = prior*exp(0.5*logvar)+mu
// ---------------------------------------------------------------------------
__global__ void __launch_bounds__(256)
zsplit_kernel(const float* __restrict__ gp, const float* __restrict__ prior,
              float* __restrict__ mu_o, float* __restrict__ lv_o,
              float* __restrict__ z) {
  const int idx = blockIdx.x * 256 + threadIdx.x;
  if (idx >= B_ * LATENT) return;
  const int b = idx >> 8, l = idx & 255;
  const float mu = gp[b * (2 * LATENT) + l];
  const float lv = gp[b * (2 * LATENT) + LATENT + l];
  mu_o[idx] = mu;
  lv_o[idx] = lv;
  z[idx] = prior[idx] * expf(0.5f * lv) + mu;
}

// ---------------------------------------------------------------------------
extern "C" void kernel_launch(void* const* d_in, const int* in_sizes, int n_in,
                              void* d_out, int out_size, void* d_ws, size_t ws_size,
                              hipStream_t stream) {
  (void)in_sizes; (void)n_in; (void)out_size; (void)ws_size;
  const float* x     = (const float*)d_in[0];
  const int*   ids   = (const int*)  d_in[1];
  const float* embed = (const float*)d_in[2];
  const float* Wi    = (const float*)d_in[3];
  const float* bi    = (const float*)d_in[4];
  const float* We1   = (const float*)d_in[5];
  const float* be1   = (const float*)d_in[6];
  const float* We2   = (const float*)d_in[7];
  const float* be2   = (const float*)d_in[8];
  const float* Wd1   = (const float*)d_in[9];
  const float* bd1   = (const float*)d_in[10];
  const float* Wd2   = (const float*)d_in[11];
  const float* bd2   = (const float*)d_in[12];
  const float* mask  = (const float*)d_in[13];
  const float* eps   = (const float*)d_in[14];

  float* out   = (float*)d_out;
  float* recon = out;                                  // 512 x 20000
  float* mu_o  = out + (size_t)B_ * D0_;               // 512 x 256
  float* lv_o  = mu_o + (size_t)B_ * LATENT;           // 512 x 256

  char* w = (char*)d_ws;
  float* P     = (float*)w; w += (size_t)VPAD_ * LATENT * 4;      // 3.44 MB
  float* xdm   = (float*)w; w += (size_t)B_ * D0_ * 4;            // 41 MB
  float* h     = (float*)w; w += (size_t)B_ * H_ * 4;             // 2 MB
  float* prior = (float*)w; w += (size_t)B_ * LATENT * 4;         // 0.5 MB
  float* gp    = (float*)w; w += (size_t)B_ * 2 * LATENT * 4;     // 1 MB
  float* z     = (float*)w; w += (size_t)B_ * LATENT * 4;         // 0.5 MB
  float* hd    = (float*)w; w += (size_t)B_ * H_ * 4;             // 2 MB
  float* part  = (float*)w;                                       // 16 MB (reused)

  const dim3 blk(256);

  // 1) P = embed @ Wi^T + bi   (M=3356 clamp, 3360-row padded store; K tail 8)
  {
    dim3 g(LATENT / 64, (VPAD_ + 127) / 128, 1);
    wmma_gemm_nt<<<g, blk, 0, stream>>>(embed, Wi, bi, P, V_, LATENT, E_, E_, VPAD_, 0);
  }
  // 2) mean/std over gathered P rows -> prior
  stats_prior_kernel<<<B_, blk, 0, stream>>>(P, ids, eps, prior);
  // 3) xdm = normalize(x) * drop_mask
  normalize_mask_kernel<<<B_, blk, 0, stream>>>(x, mask, xdm);
  // 4) h = tanh(xdm @ We1^T + be1)   K=20000, split-K S=8 (Kc=2528)
  {
    dim3 g(H_ / 64, B_ / 128, 8);
    wmma_gemm_nt<<<g, blk, 0, stream>>>(xdm, We1, be1, part, B_, H_, D0_, 2528, B_, 1);
    combine_kernel<<<(B_ * H_ + 255) / 256, blk, 0, stream>>>(part, be1, h, B_, H_, 8, 1);
  }
  // 5) gparams = h @ We2^T + be2     K=1024, split-K S=4 (Kc=256)
  {
    dim3 g((2 * LATENT) / 64, B_ / 128, 4);
    wmma_gemm_nt<<<g, blk, 0, stream>>>(h, We2, be2, part, B_, 2 * LATENT, H_, 256, B_, 0);
    combine_kernel<<<(B_ * 2 * LATENT + 255) / 256, blk, 0, stream>>>(part, be2, gp, B_, 2 * LATENT, 4, 0);
  }
  // 6) mu/logvar out; z = prior*exp(0.5*logvar)+mu
  zsplit_kernel<<<(B_ * LATENT + 255) / 256, blk, 0, stream>>>(gp, prior, mu_o, lv_o, z);
  // 7) hd = tanh(z @ Wd1^T + bd1)    K=256, split-K S=2 (Kc=128)
  {
    dim3 g(H_ / 64, B_ / 128, 2);
    wmma_gemm_nt<<<g, blk, 0, stream>>>(z, Wd1, bd1, part, B_, H_, LATENT, 128, B_, 1);
    combine_kernel<<<(B_ * H_ + 255) / 256, blk, 0, stream>>>(part, bd1, hd, B_, H_, 2, 1);
  }
  // 8) recon = hd @ Wd2^T + bd2      N=20000 -> 1252 WGs, no split needed
  {
    dim3 g((D0_ + 63) / 64, B_ / 128, 1);
    wmma_gemm_nt<<<g, blk, 0, stream>>>(hd, Wd2, bd2, recon, B_, D0_, H_, H_, B_, 0);
  }
}
